// ClusterEncoder_33758442947290
// MI455X (gfx1250) — matrix operand order repaired and checked by reading, not compiled, not run on hardware
//
#include <hip/hip_runtime.h>
#include <math.h>

// ---------------------------------------------------------------------------
// PointTransformerConv for MI455X (gfx1250, wave32, WMMA f32_16x16x32_f16)
// ---------------------------------------------------------------------------
#define NN    50000
#define EE    800000
#define CC    128      // C_IN == C_OUT
#define HH    64       // P_HID == A_HID

typedef _Float16 v16h __attribute__((ext_vector_type(16)));
typedef _Float16 v8h  __attribute__((ext_vector_type(8)));
typedef float    v8f  __attribute__((ext_vector_type(8)));

union V16H { v16h v; v8h h[2]; };

__device__ __forceinline__ v8f wmma16(v16h a, v16h b, v8f c) {
  // D = A(16x32 f16) * B(32x16 f16) + C(16x16 f32)
  return __builtin_amdgcn_wmma_f32_16x16x32_f16(
      /*neg_a=*/false, a, /*neg_b=*/false, b,
      /*c_mod=*/(short)0, c, /*reuse_a=*/false, /*reuse_b=*/false);
}

// A-operand (16x32, f16) for one k-step from a row-major f16 row pointer.
// Lane (r,hi): element i -> A[row][k0 + (i>=8?16:0) + hi*8 + (i&7)]
__device__ __forceinline__ v16h loadA(const _Float16* row, int k0, int hi) {
  V16H u;
  u.h[0] = *(const v8h*)(row + k0 + hi * 8);
  u.h[1] = *(const v8h*)(row + k0 + 16 + hi * 8);
  return u.v;
}

// B-operand (32x16, f16) from transposed weights Wt[out][in] (row stride = K).
// Lane (r,hi): element i -> Wt[col][k0 + (i>=8?16:0) + hi*8 + (i&7)]
__device__ __forceinline__ v16h loadB(const _Float16* Wt, int K, int col,
                                      int k0, int hi) {
  V16H u;
  const _Float16* p = Wt + (size_t)col * K + k0 + hi * 8;
  u.h[0] = *(const v8h*)(p);
  u.h[1] = *(const v8h*)(p + 16);
  return u.v;
}

// ---------------------------------------------------------------------------
// Weight prep: fp32 [in][out] -> f16 transposed [out][in]
// ---------------------------------------------------------------------------
__global__ void prep_weights(const float* __restrict__ Wl,
                             const float* __restrict__ Ws,
                             const float* __restrict__ Wd,
                             const float* __restrict__ Wa1,
                             const float* __restrict__ Wa2,
                             const float* __restrict__ Wp2,
                             _Float16* __restrict__ Wlt,
                             _Float16* __restrict__ Wst,
                             _Float16* __restrict__ Wdt,
                             _Float16* __restrict__ Wa1t,
                             _Float16* __restrict__ Wa2t,
                             _Float16* __restrict__ Wp2t) {
  int i = blockIdx.x * blockDim.x + threadIdx.x;
  if (i < 16384) {                       // Wl: (128,128) -> (128,128)^T
    int o = i >> 7, in = i & 127;
    Wlt[i] = (_Float16)Wl[in * 128 + o];
  } else if (i < 32768) {                // Ws
    int j = i - 16384; int o = j >> 7, in = j & 127;
    Wst[j] = (_Float16)Ws[in * 128 + o];
  } else if (i < 49152) {                // Wd
    int j = i - 32768; int o = j >> 7, in = j & 127;
    Wdt[j] = (_Float16)Wd[in * 128 + o];
  } else if (i < 57344) {                // Wa1: (128,64) -> [64][128]
    int j = i - 49152; int o = j >> 7, in = j & 127;
    Wa1t[j] = (_Float16)Wa1[in * 64 + o];
  } else if (i < 65536) {                // Wa2: (64,128) -> [128][64]
    int j = i - 57344; int o = j >> 6, in = j & 63;
    Wa2t[j] = (_Float16)Wa2[in * 128 + o];
  } else if (i < 73728) {                // Wp2: (64,128) -> [128][64]
    int j = i - 65536; int o = j >> 6, in = j & 63;
    Wp2t[j] = (_Float16)Wp2[in * 128 + o];
  }
}

__global__ void zero_f32(float* __restrict__ p, int n) {
  int i = blockIdx.x * blockDim.x + threadIdx.x;
  if (i < n) p[i] = 0.0f;
}

// ---------------------------------------------------------------------------
// Node GEMMs: h = x@Wl (f32 out), xs = x@Ws (f16), xd = x@Wd (f16)
// One wave computes one 16x128 output tile (8 N-tiles, K=128 in 4 WMMA steps).
// ---------------------------------------------------------------------------
__global__ __launch_bounds__(256) void node_gemm(
    const float* __restrict__ x,
    const _Float16* __restrict__ Wlt, const _Float16* __restrict__ Wst,
    const _Float16* __restrict__ Wdt,
    float* __restrict__ h, _Float16* __restrict__ xsh,
    _Float16* __restrict__ xdh) {
  const int lane = threadIdx.x & 31;
  const int r = lane & 15, hi = lane >> 4;
  const int gw = blockIdx.x * 8 + (threadIdx.x >> 5);
  const int nTiles = NN / 16;                 // 3125
  if (gw >= 3 * nTiles) return;
  const int mat = gw / nTiles;
  const int rowBase = (gw % nTiles) * 16;
  const _Float16* Wt = (mat == 0) ? Wlt : (mat == 1) ? Wst : Wdt;

  v8f acc[8];
#pragma unroll
  for (int n = 0; n < 8; ++n) acc[n] = (v8f)0.0f;

#pragma unroll
  for (int ks = 0; ks < 4; ++ks) {
    const int k0 = ks * 32;
    const float* arow = x + (size_t)(rowBase + r) * CC + k0;
    V16H a;
#pragma unroll
    for (int i = 0; i < 8; ++i) {
      a.v[i]     = (_Float16)arow[hi * 8 + i];
      a.v[8 + i] = (_Float16)arow[16 + hi * 8 + i];
    }
#pragma unroll
    for (int n = 0; n < 8; ++n)
      acc[n] = wmma16(a.v, loadB(Wt, 128, n * 16 + r, k0, hi), acc[n]);
  }

#pragma unroll
  for (int n = 0; n < 8; ++n) {
    const int col = n * 16 + r;
#pragma unroll
    for (int g = 0; g < 8; ++g) {
      const size_t row = rowBase + g + hi * 8;
      if (mat == 0)      h[row * CC + col]   = acc[n][g];
      else if (mat == 1) xsh[row * CC + col] = (_Float16)acc[n][g];
      else               xdh[row * CC + col] = (_Float16)acc[n][g];
    }
  }
}

// ---------------------------------------------------------------------------
// Edge pipeline: one wave per 32-edge tile (two 16-row WMMA sub-tiles so each
// loaded B operand feeds two WMMAs); 2 waves per block.
//   G : a = xd[dst]-xs[src]            -> sT      (f16, LDS)
//   P1: ph = relu(posdiff@Wp1+bp1)     -> sPU     (VALU, K=2)
//   P2: delta = relu(ph@Wp2+bp2)       -> sDelta; sT += delta   (WMMA)
//   A1: u = relu(sT@Wa1+ba1)           -> sPU     (WMMA)
//   A2: logit = relu(u@Wa2+ba2)        -> regs    (WMMA)
//   PASS 1: atomicMax(m);  PASS 2: e=exp(l-m); atomicAdd(s, acc)
// ---------------------------------------------------------------------------
template <int PASS>
__global__ __launch_bounds__(64) void edge_pass(
    const int* __restrict__ edges, const float* __restrict__ pos,
    const _Float16* __restrict__ xsh, const _Float16* __restrict__ xdh,
    const float* __restrict__ h,
    const _Float16* __restrict__ Wp2t, const _Float16* __restrict__ Wa1t,
    const _Float16* __restrict__ Wa2t,
    const float* __restrict__ Wp1, const float* __restrict__ bp1,
    const float* __restrict__ bp2, const float* __restrict__ ba1,
    const float* __restrict__ ba2,
    float* __restrict__ m, float* __restrict__ sden,
    float* __restrict__ accv) {
  __shared__ _Float16 sT[2][32][CC];      // 16 KB  (a, then a+delta)
  __shared__ _Float16 sDelta[2][32][CC];  // 16 KB
  __shared__ _Float16 sPU[2][32][HH];     // 8 KB   (ph, then u)
  __shared__ int sSrc[2][32], sDst[2][32];
  __shared__ float sWp1[128], sBp1[64];

  const int tid = threadIdx.x;
  const int w = tid >> 5, lane = tid & 31;
  const int r = lane & 15, hi = lane >> 4;
  const int tile = blockIdx.x * 2 + w;
  const int eBase = tile * 32;

  sWp1[tid] = Wp1[tid];
  sWp1[64 + tid] = Wp1[64 + tid];
  sBp1[tid] = bp1[tid];

  // --- stage 0 + G: indices, pos-diff, gather a = xd[dst]-xs[src]
  float p0[2], p1[2];
#pragma unroll
  for (int t = 0; t < 2; ++t) {
    const int e  = eBase + t * 16 + r;
    const int si = edges[e];
    const int di = edges[EE + e];
    if (lane < 16) { sSrc[w][t * 16 + lane] = si; sDst[w][t * 16 + lane] = di; }
    p0[t] = pos[di * 2]     - pos[si * 2];
    p1[t] = pos[di * 2 + 1] - pos[si * 2 + 1];
    const _Float16* rd = xdh + (size_t)di * CC + hi * 64;
    const _Float16* rs = xsh + (size_t)si * CC + hi * 64;
#pragma unroll
    for (int c = 0; c < 8; ++c) {
      v8h ad = *(const v8h*)(rd + c * 8);
      v8h as = *(const v8h*)(rs + c * 8);
      *(v8h*)&sT[w][t * 16 + r][hi * 64 + c * 8] = ad - as;
    }
  }
  __syncthreads();

  // --- P1 (K=2, VALU): lane covers rows {r, 16+r}, hidden cols hi*32..+31
#pragma unroll
  for (int t = 0; t < 2; ++t) {
#pragma unroll 8
    for (int p = 0; p < 32; ++p) {
      const int pp = hi * 32 + p;
      float v = fmaf(p0[t], sWp1[pp], fmaf(p1[t], sWp1[64 + pp], sBp1[pp]));
      sPU[w][t * 16 + r][pp] = (_Float16)fmaxf(v, 0.0f);
    }
  }
  __syncthreads();

  // --- P2: delta = relu(ph @ Wp2t + bp2)   (K=64, 8 N-tiles, 2 M-tiles)
  {
    v8f d[2][8];
#pragma unroll
    for (int t = 0; t < 2; ++t)
#pragma unroll
      for (int n = 0; n < 8; ++n) d[t][n] = (v8f)0.0f;
#pragma unroll
    for (int ks = 0; ks < 2; ++ks) {
      const int k0 = ks * 32;
      v16h a0 = loadA(&sPU[w][r][0], k0, hi);
      v16h a1 = loadA(&sPU[w][16 + r][0], k0, hi);
#pragma unroll
      for (int n = 0; n < 8; ++n) {
        v16h b = loadB(Wp2t, 64, n * 16 + r, k0, hi);
        d[0][n] = wmma16(a0, b, d[0][n]);
        d[1][n] = wmma16(a1, b, d[1][n]);
      }
    }
#pragma unroll
    for (int n = 0; n < 8; ++n) {
      const int col = n * 16 + r;
      const float bias = bp2[col];
#pragma unroll
      for (int t = 0; t < 2; ++t) {
#pragma unroll
        for (int g = 0; g < 8; ++g) {
          const int row = t * 16 + g + hi * 8;
          _Float16 hv = (_Float16)fmaxf(d[t][n][g] + bias, 0.0f);
          sDelta[w][row][col] = hv;
          sT[w][row][col] = sT[w][row][col] + hv;   // t = a + delta
        }
      }
    }
  }
  __syncthreads();

  // --- A1: u = relu(t @ Wa1t + ba1)   (K=128, 4 N-tiles, 2 M-tiles)
  {
    v8f u[2][4];
#pragma unroll
    for (int t = 0; t < 2; ++t)
#pragma unroll
      for (int n = 0; n < 4; ++n) u[t][n] = (v8f)0.0f;
#pragma unroll
    for (int ks = 0; ks < 4; ++ks) {
      const int k0 = ks * 32;
      v16h a0 = loadA(&sT[w][r][0], k0, hi);
      v16h a1 = loadA(&sT[w][16 + r][0], k0, hi);
#pragma unroll
      for (int n = 0; n < 4; ++n) {
        v16h b = loadB(Wa1t, 128, n * 16 + r, k0, hi);
        u[0][n] = wmma16(a0, b, u[0][n]);
        u[1][n] = wmma16(a1, b, u[1][n]);
      }
    }
    // sPU (ph) was fully consumed in P2 (barrier passed): overwrite with u
#pragma unroll
    for (int n = 0; n < 4; ++n) {
      const int col = n * 16 + r;
      const float bias = ba1[col];
#pragma unroll
      for (int t = 0; t < 2; ++t)
#pragma unroll
        for (int g = 0; g < 8; ++g)
          sPU[w][t * 16 + g + hi * 8][col] =
              (_Float16)fmaxf(u[t][n][g] + bias, 0.0f);
    }
  }
  __syncthreads();

  // --- A2: logit = relu(u @ Wa2t + ba2)  (K=64, 8 N-tiles, 2 M-tiles) -> regs
  v8f lg[2][8];
#pragma unroll
  for (int t = 0; t < 2; ++t)
#pragma unroll
    for (int n = 0; n < 8; ++n) lg[t][n] = (v8f)0.0f;
#pragma unroll
  for (int ks = 0; ks < 2; ++ks) {
    const int k0 = ks * 32;
    v16h a0 = loadA(&sPU[w][r][0], k0, hi);
    v16h a1 = loadA(&sPU[w][16 + r][0], k0, hi);
#pragma unroll
    for (int n = 0; n < 8; ++n) {
      v16h b = loadB(Wa2t, 64, n * 16 + r, k0, hi);
      lg[0][n] = wmma16(a0, b, lg[0][n]);
      lg[1][n] = wmma16(a1, b, lg[1][n]);
    }
  }

  // --- epilogue: softmax passes
#pragma unroll
  for (int t = 0; t < 2; ++t) {
    int dstr[8], srcr[8];
#pragma unroll
    for (int g = 0; g < 8; ++g) {
      dstr[g] = sDst[w][t * 16 + g + hi * 8];
      srcr[g] = sSrc[w][t * 16 + g + hi * 8];
    }
#pragma unroll
    for (int n = 0; n < 8; ++n) {
      const int col = n * 16 + r;
      const float bias = ba2[col];
#pragma unroll
      for (int g = 0; g < 8; ++g) {
        const float lv = fmaxf(lg[t][n][g] + bias, 0.0f);   // logit >= 0
        const size_t off = (size_t)dstr[g] * CC + col;
        if (PASS == 1) {
          atomicMax((int*)m + off, __float_as_int(lv));  // valid: lv >= 0
        } else {
          const float ev = __expf(lv - m[off]);
          atomicAdd(sden + off, ev);
          const float hv = h[(size_t)srcr[g] * CC + col];
          const float dv = (float)sDelta[w][t * 16 + g + hi * 8][col];
          atomicAdd(accv + off, ev * (hv + dv));
        }
      }
    }
  }
}

__global__ void finalize(const float* __restrict__ accv,
                         const float* __restrict__ sden,
                         float* __restrict__ out, int n) {
  int i = blockIdx.x * blockDim.x + threadIdx.x;
  if (i < n) out[i] = fmaxf(accv[i] / (sden[i] + 1e-16f), 0.0f);
}

// ---------------------------------------------------------------------------
extern "C" void kernel_launch(void* const* d_in, const int* in_sizes, int n_in,
                              void* d_out, int out_size, void* d_ws,
                              size_t ws_size, hipStream_t stream) {
  const float* x    = (const float*)d_in[0];
  const float* pos  = (const float*)d_in[1];
  const int*   ei   = (const int*)d_in[2];
  const float* Wl   = (const float*)d_in[3];
  const float* Ws   = (const float*)d_in[4];
  const float* Wd   = (const float*)d_in[5];
  const float* Wp1  = (const float*)d_in[6];
  const float* bp1  = (const float*)d_in[7];
  const float* Wp2  = (const float*)d_in[8];
  const float* bp2  = (const float*)d_in[9];
  const float* Wa1  = (const float*)d_in[10];
  const float* ba1  = (const float*)d_in[11];
  const float* Wa2  = (const float*)d_in[12];
  const float* ba2  = (const float*)d_in[13];
  float* out = (float*)d_out;

  // workspace layout (all chunks 256B-multiple sized)
  char* p = (char*)d_ws;
  _Float16* Wlt  = (_Float16*)p; p += 128 * 128 * 2;
  _Float16* Wst  = (_Float16*)p; p += 128 * 128 * 2;
  _Float16* Wdt  = (_Float16*)p; p += 128 * 128 * 2;
  _Float16* Wa1t = (_Float16*)p; p += 64 * 128 * 2;
  _Float16* Wa2t = (_Float16*)p; p += 128 * 64 * 2;
  _Float16* Wp2t = (_Float16*)p; p += 128 * 64 * 2;
  float*    h    = (float*)p;    p += (size_t)NN * CC * 4;
  _Float16* xsh  = (_Float16*)p; p += (size_t)NN * CC * 2;
  _Float16* xdh  = (_Float16*)p; p += (size_t)NN * CC * 2;
  float*    m    = (float*)p;    p += (size_t)NN * CC * 4;
  float*    sden = (float*)p;    p += (size_t)NN * CC * 4;
  float*    accv = (float*)p;    p += (size_t)NN * CC * 4;

  // 1) weight transpose+convert
  prep_weights<<<(73728 + 255) / 256, 256, 0, stream>>>(
      Wl, Ws, Wd, Wa1, Wa2, Wp2, Wlt, Wst, Wdt, Wa1t, Wa2t, Wp2t);

  // 2) zero softmax accumulators (m, sden, accv are contiguous)
  const int zc = 3 * NN * CC;
  zero_f32<<<(zc + 255) / 256, 256, 0, stream>>>(m, zc);

  // 3) node GEMMs (3 output matrices * 3125 row-tiles, 8 waves/block)
  const int nodeWaves = 3 * (NN / 16);
  node_gemm<<<(nodeWaves + 7) / 8, 256, 0, stream>>>(x, Wlt, Wst, Wdt, h, xsh,
                                                     xdh);

  // 4/5) edge pipeline, two passes (25000 32-edge tiles, 2 waves/block)
  const int edgeBlocks = (EE / 32) / 2;
  edge_pass<1><<<edgeBlocks, 64, 0, stream>>>(ei, pos, xsh, xdh, h, Wp2t,
                                              Wa1t, Wa2t, Wp1, bp1, bp2, ba1,
                                              ba2, m, sden, accv);
  edge_pass<2><<<edgeBlocks, 64, 0, stream>>>(ei, pos, xsh, xdh, h, Wp2t,
                                              Wa1t, Wa2t, Wp1, bp1, bp2, ba1,
                                              ba2, m, sden, accv);

  // 6) out = relu(acc / (s + eps))
  const int oc = NN * CC;
  finalize<<<(oc + 255) / 256, 256, 0, stream>>>(accv, sden, out, oc);
}
